// STGCNBlock_17721035063789
// MI455X (gfx1250) — matrix operand order repaired and checked by reading, not compiled, not run on hardware
//
#include <hip/hip_runtime.h>

// ---------------------------------------------------------------------------
// Problem constants
// ---------------------------------------------------------------------------
#define BATCH 32
#define NNODE 512
#define TT    24
#define FIN   3
#define HID   128
#define FOUT  64
#define CCOL  (TT * FOUT)           // 1536 flattened (t,f) columns
#define ZELEMS (25165824ull)        // B*N*T*F
#define ODE_STEPS 6

typedef __attribute__((ext_vector_type(16))) __bf16 v16bf;
typedef __attribute__((ext_vector_type(8)))  __bf16 v8bf;
typedef __attribute__((ext_vector_type(8)))  float  v8f;

__device__ inline v8f wmma_bf16(v16bf a, v16bf b, v8f c) {
  // D = A(16x32 bf16) * B(32x16 bf16) + C(16x16 f32)
  return __builtin_amdgcn_wmma_f32_16x16x32_bf16(
      false, a, false, b, (short)0, c, false, false);
}

// ---------------------------------------------------------------------------
// Prep: A_hat -> bf16, sigmoid(alpha), w_eff (transposed, bf16), w2_eff (f32)
// ---------------------------------------------------------------------------
__global__ __launch_bounds__(256)
void prep_kernel(const float* __restrict__ A_hat, const float* __restrict__ alpha,
                 const float* __restrict__ w,  const float* __restrict__ d,
                 const float* __restrict__ w2, const float* __restrict__ d2,
                 __bf16* __restrict__ Abf, __bf16* __restrict__ weT,
                 float* __restrict__ w2e, float* __restrict__ sa) {
  int gid = blockIdx.x * 256 + threadIdx.x;
  if (gid < NNODE * NNODE) Abf[gid] = (__bf16)A_hat[gid];
  if (gid < NNODE) sa[gid] = 1.0f / (1.0f + expf(-alpha[gid]));
  if (gid < TT * TT) {                 // w2_eff[ti][t_out]
    int ti = gid / TT, m = gid % TT;
    float s = 0.0f;
    for (int k = 0; k < TT; ++k) {
      float dk = fminf(fmaxf(d2[k], 0.0f), 1.0f);
      s = fmaf(w2[ti * TT + k] * dk, w2[m * TT + k], s);
    }
    w2e[gid] = s;
  }
  if (gid < FOUT * FOUT) {             // weT[g_out*64 + f] = w_eff[f][g_out]
    int go = gid >> 6, ff = gid & 63;
    float s = 0.0f;
    for (int k = 0; k < FOUT; ++k) {
      float dk = fminf(fmaxf(d[k], 0.0f), 1.0f);
      s = fmaf(w[ff * FOUT + k] * dk, w[go * FOUT + k], s);
    }
    weT[gid] = (__bf16)s;
  }
}

// ---------------------------------------------------------------------------
// Pointwise MLP: h = relu(X@W1+b1)@W2+b2 ; writes z-ping and x0 (== d_out)
// ---------------------------------------------------------------------------
__global__ __launch_bounds__(256)
void mlp_kernel(const float* __restrict__ X,
                const float* __restrict__ W1, const float* __restrict__ b1,
                const float* __restrict__ W2, const float* __restrict__ b2,
                float* __restrict__ z0, float* __restrict__ x0) {
  __shared__ float hid[4][HID];
  const size_t row0 = (size_t)blockIdx.x * 4;           // rows of B*N*T
  for (int idx = threadIdx.x; idx < 4 * HID; idx += 256) {
    int lr = idx >> 7, hk = idx & 127;
    const float* xr = X + (row0 + lr) * FIN;
    float h = b1[hk];
    h = fmaf(xr[0], W1[0 * HID + hk], h);
    h = fmaf(xr[1], W1[1 * HID + hk], h);
    h = fmaf(xr[2], W1[2 * HID + hk], h);
    hid[lr][hk] = fmaxf(h, 0.0f);
  }
  __syncthreads();
  for (int idx = threadIdx.x; idx < 4 * FOUT; idx += 256) {
    int lr = idx >> 6, fo = idx & 63;
    float s = b2[fo];
#pragma unroll 8
    for (int k = 0; k < HID; ++k) s = fmaf(hid[lr][k], W2[k * FOUT + fo], s);
    size_t o = (row0 + lr) * FOUT + fo;
    z0[o] = s;
    x0[o] = s;
  }
}

// ---------------------------------------------------------------------------
// LDS-tiled transpose + convert: z (f32, [b][n][c]) -> zT (bf16, [b][c][n]).
// Both global read (contiguous c) and global write (contiguous n) coalesced.
// ---------------------------------------------------------------------------
__global__ __launch_bounds__(256)
void cvt_kernel(const float* __restrict__ z, __bf16* __restrict__ zT) {
  __shared__ float tile[32][33];                  // +1 pad: no LDS bank conflicts
  const int b  = blockIdx.z;
  const int n0 = blockIdx.y * 32;
  const int c0 = blockIdx.x * 32;
  const int cx = threadIdx.x & 31;                // fast-varying lane index
  const int ry = threadIdx.x >> 5;                // 0..7
#pragma unroll
  for (int i = 0; i < 4; ++i) {
    int n = ry + i * 8;
    tile[n][cx] = z[((size_t)b * NNODE + n0 + n) * CCOL + c0 + cx];
  }
  __syncthreads();
#pragma unroll
  for (int i = 0; i < 4; ++i) {
    int cc = ry + i * 8;
    zT[((size_t)b * CCOL + c0 + cc) * NNODE + n0 + cx] = (__bf16)tile[cx][cc];
  }
}

// ---------------------------------------------------------------------------
// One Euler step: z_next = 0.5*sig(alpha)*(A@z) + z@w_eff + temporal(z) + x0 - 2z
// One wave computes a 16(node) x 32(col) tile (two 16x16 WMMA tiles sharing the
// A fragment). 8 waves/block cover 256 columns.
// ---------------------------------------------------------------------------
__global__ __launch_bounds__(256)
void step_kernel(const float* __restrict__ z, const __bf16* __restrict__ zT,
                 const __bf16* __restrict__ Abf, const __bf16* __restrict__ weT,
                 const float* __restrict__ w2e, const float* __restrict__ sa,
                 const float* __restrict__ x0, float* __restrict__ zn) {
  const int lane = threadIdx.x & 31;
  const int ln = lane & 15;          // column / row-in-halfwave
  const int lg = lane >> 4;          // half-wave group
  const int wv = threadIdx.x >> 5;
  const int c0 = blockIdx.x * 256 + wv * 32;   // two col tiles: c0, c0+16
  const int i0 = blockIdx.y * 16;              // node tile base
  const int b  = blockIdx.z;
  const int t  = c0 / FOUT;                    // 32-col tile lies in one t-block
  const int f0 = c0 % FOUT;                    // 0 or 32

  // ---- graph diffusion: A_hat[i0:i0+16, :] @ z_b[:, c0:c0+32]  (K = 512)
  v8f acc0 = {0.f,0.f,0.f,0.f,0.f,0.f,0.f,0.f};
  v8f acc1 = {0.f,0.f,0.f,0.f,0.f,0.f,0.f,0.f};
  {
    const __bf16* Abase = Abf + (size_t)(i0 + ln) * NNODE + lg * 16;
    const __bf16* B0 = zT + ((size_t)b * CCOL + (c0 + ln)) * NNODE + lg * 16;
    const __bf16* B1 = B0 + (size_t)16 * NNODE;
#pragma unroll 2
    for (int k0 = 0; k0 < NNODE; k0 += 32) {
      v8bf alo = *(const v8bf*)(Abase + k0);
      v8bf ahi = *(const v8bf*)(Abase + k0 + 8);
      v8bf b0l = *(const v8bf*)(B0 + k0);
      v8bf b0h = *(const v8bf*)(B0 + k0 + 8);
      v8bf b1l = *(const v8bf*)(B1 + k0);
      v8bf b1h = *(const v8bf*)(B1 + k0 + 8);
      v16bf af, bf0, bf1;
#pragma unroll
      for (int e = 0; e < 8; ++e) {
        af[e] = alo[e];  af[e + 8]  = ahi[e];
        bf0[e] = b0l[e]; bf0[e + 8] = b0h[e];
        bf1[e] = b1l[e]; bf1[e + 8] = b1h[e];
      }
      acc0 = wmma_bf16(af, bf0, acc0);
      acc1 = wmma_bf16(af, bf1, acc1);
    }
  }

  // ---- feature mix: z_b[i0:i0+16, t, :] @ w_eff[:, f0:f0+32]  (K = 64)
  v8f accw0 = {0.f,0.f,0.f,0.f,0.f,0.f,0.f,0.f};
  v8f accw1 = {0.f,0.f,0.f,0.f,0.f,0.f,0.f,0.f};
  {
    const float*  zrow = z + ((size_t)b * NNODE + i0 + ln) * CCOL + t * FOUT + lg * 16;
    const __bf16* wc0 = weT + (size_t)(f0 + ln) * FOUT + lg * 16;
    const __bf16* wc1 = wc0 + (size_t)16 * FOUT;
#pragma unroll
    for (int kk = 0; kk < FOUT; kk += 32) {
      v16bf af, bf0, bf1;
#pragma unroll
      for (int e = 0; e < 8; ++e) {
        af[e]     = (__bf16)zrow[kk + e];
        af[e + 8] = (__bf16)zrow[kk + 8 + e];
      }
#pragma unroll
      for (int e = 0; e < 16; ++e) { bf0[e] = wc0[kk + e]; bf1[e] = wc1[kk + e]; }
      accw0 = wmma_bf16(af, bf0, accw0);
      accw1 = wmma_bf16(af, bf1, accw1);
    }
  }

  // preload temporal-mix weights for this t (uniform per wave)
  float w2r[TT];
#pragma unroll
  for (int m = 0; m < TT; ++m) w2r[m] = w2e[m * TT + t];

  const int cA = c0 + ln;
  const int fA = f0 + ln;
#pragma unroll
  for (int r = 0; r < 8; ++r) {
    const int n = i0 + r + 8 * lg;                 // C/D layout: VGPR r -> M = r+8*lg
    const float sc = 0.5f * sa[n];
    const size_t base = ((size_t)b * NNODE + n) * CCOL;
    float zc0  = z[base + cA];
    float zc1  = z[base + cA + 16];
    float x00  = x0[base + cA];
    float x01  = x0[base + cA + 16];
    float t0 = 0.0f, t1 = 0.0f;                    // temporal mix, K = 24 (VALU)
#pragma unroll
    for (int m = 0; m < TT; ++m) {
      const float* zm = z + base + m * FOUT;
      t0 = fmaf(zm[fA],      w2r[m], t0);
      t1 = fmaf(zm[fA + 16], w2r[m], t1);
    }
    zn[base + cA]      = fmaf(sc, acc0[r], accw0[r] + t0 + x00 - 2.0f * zc0);
    zn[base + cA + 16] = fmaf(sc, acc1[r], accw1[r] + t1 + x01 - 2.0f * zc1);
  }
}

// ---------------------------------------------------------------------------
// BatchNorm over channel = node: per-node mean / inv-std of relu(z)
// ---------------------------------------------------------------------------
__global__ __launch_bounds__(256)
void stats_kernel(const float* __restrict__ z, float* __restrict__ mean,
                  float* __restrict__ istd) {
  const int n = blockIdx.x;
  float s = 0.0f, s2 = 0.0f;
  for (int idx = threadIdx.x; idx < BATCH * CCOL; idx += 256) {
    int bb = idx / CCOL, c = idx % CCOL;
    float v = fmaxf(z[((size_t)bb * NNODE + n) * CCOL + c], 0.0f);
    s += v;
    s2 = fmaf(v, v, s2);
  }
  __shared__ float sh[256], sh2[256];
  sh[threadIdx.x] = s; sh2[threadIdx.x] = s2;
  __syncthreads();
  for (int st = 128; st > 0; st >>= 1) {
    if (threadIdx.x < st) {
      sh[threadIdx.x]  += sh[threadIdx.x + st];
      sh2[threadIdx.x] += sh2[threadIdx.x + st];
    }
    __syncthreads();
  }
  if (threadIdx.x == 0) {
    const float inv = 1.0f / (float)(BATCH * CCOL);
    float m = sh[0] * inv;
    float var = sh2[0] * inv - m * m;
    mean[n] = m;
    istd[n] = rsqrtf(var + 1e-5f);
  }
}

__global__ __launch_bounds__(256)
void apply_kernel(const float* __restrict__ z, const float* __restrict__ mean,
                  const float* __restrict__ istd, const float* __restrict__ gamma,
                  const float* __restrict__ beta, float* __restrict__ out) {
  size_t gid = (size_t)blockIdx.x * 256 + threadIdx.x;
  if (gid >= ZELEMS) return;
  int n = (int)((gid / CCOL) & (NNODE - 1));
  float v = fmaxf(z[gid], 0.0f);
  out[gid] = fmaf((v - mean[n]) * istd[n], gamma[n], beta[n]);
}

// ---------------------------------------------------------------------------
// Launcher
// ---------------------------------------------------------------------------
extern "C" void kernel_launch(void* const* d_in, const int* in_sizes, int n_in,
                              void* d_out, int out_size, void* d_ws, size_t ws_size,
                              hipStream_t stream) {
  const float* X     = (const float*)d_in[0];
  const float* A_hat = (const float*)d_in[1];
  const float* W1    = (const float*)d_in[2];
  const float* b1    = (const float*)d_in[3];
  const float* W2    = (const float*)d_in[4];
  const float* b2    = (const float*)d_in[5];
  const float* alpha = (const float*)d_in[6];
  const float* w     = (const float*)d_in[7];
  const float* d     = (const float*)d_in[8];
  const float* w2    = (const float*)d_in[9];
  const float* d2    = (const float*)d_in[10];
  const float* gamma = (const float*)d_in[11];
  const float* beta  = (const float*)d_in[12];
  float* out = (float*)d_out;

  char* ws = (char*)d_ws;
  size_t off = 0;
  auto carve = [&](size_t bytes) -> char* {
    char* p = ws + off;
    off += (bytes + 255) & ~(size_t)255;
    return p;
  };

  float*  z0  = (float*) carve(ZELEMS * 4);        // 100.7 MB
  float*  z1  = (float*) carve(ZELEMS * 4);        // 100.7 MB
  __bf16* zT  = (__bf16*)carve(ZELEMS * 2);        //  50.3 MB
  __bf16* Abf = (__bf16*)carve((size_t)NNODE * NNODE * 2);
  __bf16* weT = (__bf16*)carve((size_t)FOUT * FOUT * 2);
  float*  w2e = (float*) carve((size_t)TT * TT * 4);
  float*  sa  = (float*) carve((size_t)NNODE * 4);
  float*  mn  = (float*) carve((size_t)NNODE * 4);
  float*  isd = (float*) carve((size_t)NNODE * 4);

  // x0 (the detached MLP output) lives in d_out until the final BN overwrite.
  prep_kernel<<<(NNODE * NNODE + 255) / 256, 256, 0, stream>>>(
      A_hat, alpha, w, d, w2, d2, Abf, weT, w2e, sa);
  mlp_kernel<<<(BATCH * NNODE * TT) / 4, 256, 0, stream>>>(
      X, W1, b1, W2, b2, z0, out);

  for (int s = 0; s < ODE_STEPS; ++s) {
    float* zc = (s & 1) ? z1 : z0;
    float* zn = (s & 1) ? z0 : z1;
    cvt_kernel<<<dim3(CCOL / 32, NNODE / 32, BATCH), 256, 0, stream>>>(zc, zT);
    step_kernel<<<dim3(CCOL / 256, NNODE / 16, BATCH), 256, 0, stream>>>(
        zc, zT, Abf, weT, w2e, sa, out, zn);
  }
  // 6 steps -> final state back in z0
  stats_kernel<<<NNODE, 256, 0, stream>>>(z0, mn, isd);
  apply_kernel<<<(unsigned)((ZELEMS + 255) / 256), 256, 0, stream>>>(
      z0, mn, isd, gamma, beta, out);
}
